// StandardAttentionWithHilbert_69776038690879
// MI455X (gfx1250) — matrix operand order repaired and checked by reading, not compiled, not run on hardware
//
#include <hip/hip_runtime.h>
#include <math.h>

#define NTOK 4096
#define CDIM 1024
#define QKV_DIM 3072
#define NHEADS 16
#define HDIM 64
#define ATT_SCALE 0.125f   // 64^-0.5

// Prefer the Tensor Data Mover (TENSORcnt) when the builtin exists; otherwise
// fall back to per-lane async global->LDS b128 copies (ASYNCcnt) which were
// verified to assemble in an earlier round.
#if __has_builtin(__builtin_amdgcn_tensor_load_to_lds) && \
    __has_builtin(__builtin_amdgcn_s_wait_tensorcnt)
#define USE_TDM 1
#else
#define USE_TDM 0
#endif

typedef __attribute__((ext_vector_type(16))) _Float16 v16h;
typedef __attribute__((ext_vector_type(8)))  float    v8f;
typedef __attribute__((ext_vector_type(4)))  unsigned u32x4;
typedef __attribute__((ext_vector_type(8)))  int      i32x8;
typedef __attribute__((ext_vector_type(4)))  int      i32x4;

__device__ __forceinline__ v8f wmma32(v16h a, v16h b, v8f c) {
  return __builtin_amdgcn_wmma_f32_16x16x32_f16(false, a, false, b, (short)0, c,
                                                false, false);
}

// A-fragment: 16x32 (MxK) f16, M on lanes, K packed 2-per-VGPR (ISA 7.12.2).
__device__ __forceinline__ v16h load_a_frag(const _Float16* base, int ld,
                                            int row0, int k0) {
  const int lane = threadIdx.x & 31;
  const int m = row0 + (lane & 15);
  const int g = (lane >> 4) & 1;
  const _Float16* p = base + (size_t)m * ld + k0;
  v16h a;
#pragma unroll
  for (int v = 0; v < 8; ++v) {
    const int k = g * 16 + v * 2;
    a[2 * v]     = p[k];
    a[2 * v + 1] = p[k + 1];
  }
  return a;
}

// B-fragment (KxN) from a row-major matrix whose ROWS are the N dim
// (B = W^T tile): element(k,n) = base[(col0+n)*ld + k0 + k]. Contiguous
// 16-half read per lane -> b128 (global) / ds_load (LDS).
__device__ __forceinline__ v16h load_bT_frag(const _Float16* base, int ld,
                                             int col0, int k0) {
  const int lane = threadIdx.x & 31;
  const int n = lane & 15;
  const int g = (lane >> 4) & 1;
  const _Float16* p = base + (size_t)(col0 + n) * ld + k0 + g * 16;
  v16h b;
#pragma unroll
  for (int v = 0; v < 8; ++v) {
    b[2 * v]     = p[2 * v];
    b[2 * v + 1] = p[2 * v + 1];
  }
  return b;
}

// ---- CDNA5 staging: TDM 2D tile load (preferred) or async b128 fallback ---
#if USE_TDM
// D# per CDNA5 ISA: group0 = {flags, lds_addr, global_addr,
// type=2}; group1 = {data_size=2B, tensor dims, tile dims, dim0 stride}.
// Loads a tile_d1 x tile_d0 (rows x elems) f16 tile whose rows are
// `stride0` elements apart in memory into contiguous LDS at lds_addr.
// This toolchain declares the 6-arg builtin:
//   (u32x4 g0, i32x8 g1, i32x4 g2, i32x4 g3, i32x8 g4, i32 cpol)
__device__ __forceinline__ void tdm_load_2d(unsigned lds_addr,
                                            const void* gaddr,
                                            unsigned tensor_d0,
                                            unsigned tensor_d1,
                                            unsigned tile_d0,
                                            unsigned tile_d1,
                                            unsigned long long stride0) {
  const unsigned long long ga = (unsigned long long)gaddr;
  u32x4 g0 = {};
  g0[0] = 1u;                                           // count=1
  g0[1] = lds_addr;                                     // LDS byte address
  g0[2] = (unsigned)(ga & 0xffffffffu);                 // global_addr[31:0]
  g0[3] = (unsigned)((ga >> 32) & 0x01ffffffu) | (2u << 30);  // [56:32]|type=2
  i32x8 g1 = {};
  g1[0] = (int)(1u << 16);                              // data_size=2B
  g1[1] = (int)((tensor_d0 & 0xffffu) << 16);           // tensor_dim0[15:0]
  g1[2] = (int)(((tensor_d0 >> 16) & 0xffffu) |
                ((tensor_d1 & 0xffffu) << 16));         // td0[31:16]|td1[15:0]
  g1[3] = (int)(((tensor_d1 >> 16) & 0xffffu) |
                ((tile_d0 & 0xffffu) << 16));           // td1[31:16]|tile_d0
  g1[4] = (int)(tile_d1 & 0xffffu);                     // tile_d1, tile_d2=0
  g1[5] = (int)(stride0 & 0xffffffffu);                 // stride0[31:0]
  g1[6] = (int)((stride0 >> 32) & 0xffffu);             // stride0[47:32]
  g1[7] = 0;
  i32x4 g2 = {}, g3 = {};
  i32x8 g4 = {};
  __builtin_amdgcn_tensor_load_to_lds(g0, g1, g2, g3, g4, 0);
}
#else
__device__ __forceinline__ void stage_b128(_Float16* lds_dst,
                                           const _Float16* gsrc) {
  const unsigned int loff = (unsigned int)(uintptr_t)lds_dst;
  asm volatile("global_load_async_to_lds_b128 %0, %1, off"
               :: "v"(loff), "v"(gsrc)
               : "memory");
}
__device__ __forceinline__ void wait_async0() {
#if __has_builtin(__builtin_amdgcn_s_wait_asynccnt)
  __builtin_amdgcn_s_wait_asynccnt(0);
#else
  asm volatile("s_wait_asynccnt 0x0" ::: "memory");
#endif
}
#endif

// ---------------------------------------------------------------- utilities
__global__ void cvt_f32_to_f16(const float* __restrict__ in,
                               _Float16* __restrict__ out, int n) {
  int i = blockIdx.x * blockDim.x + threadIdx.x;
  if (i < n) out[i] = (_Float16)in[i];
}

__global__ void build_inv(const int* __restrict__ idx, int* __restrict__ inv,
                          int n) {
  int i = blockIdx.x * blockDim.x + threadIdx.x;
  if (i < n) inv[idx[i]] = i;
}

// ------------------------------------------------ QKV GEMM + Hilbert gather
// qkv[n,d] = sum_c x[n,c] * w_qkv[d,c]; scatter row n to sorted slot inv[n].
// q: [head][N][64] pre-scaled by 1/sqrt(d);  k: [head][N][64];
// v: stored TRANSPOSED as [head][64][N].
__global__ void __launch_bounds__(32)
qkv_gemm(const _Float16* __restrict__ xh, const _Float16* __restrict__ wh,
         const int* __restrict__ inv, _Float16* __restrict__ q,
         _Float16* __restrict__ k, _Float16* __restrict__ vt) {
  const int d0 = blockIdx.x * 16;  // 0..3071
  const int t0 = blockIdx.y * 16;  // token tile
  v8f acc = {};
#pragma unroll 2
  for (int kk = 0; kk < CDIM; kk += 32) {
    v16h a = load_a_frag(xh, CDIM, t0, kk);
    v16h b = load_bT_frag(wh, CDIM, d0, kk);
    acc = wmma32(a, b, acc);
  }
  const int lane = threadIdx.x & 31;
  const int g = (lane >> 4) & 1;
  const int d = d0 + (lane & 15);
  const int which = d >> 10;       // 0:q 1:k 2:v (uniform per block)
  const int dd = d & (CDIM - 1);
  const int head = dd >> 6;
  const int hd = dd & (HDIM - 1);
  // hoist the uniform which-branch OUT of the store loops (straight-line SALU)
  if (which == 2) {
    _Float16* dst = vt + ((size_t)head * HDIM + hd) * NTOK;
#pragma unroll
    for (int r = 0; r < 8; ++r) {
      const int n = t0 + r + 8 * g;
      dst[inv[n]] = (_Float16)acc[r];
    }
  } else {
    const float sc = (which == 0) ? ATT_SCALE : 1.0f;
    _Float16* dst = ((which == 0) ? q : k) + (size_t)head * NTOK * HDIM + hd;
#pragma unroll
    for (int r = 0; r < 8; ++r) {
      const int n = t0 + r + 8 * g;
      dst[(size_t)inv[n] * HDIM] = (_Float16)(acc[r] * sc);
    }
  }
}

// --------------------------------------------------------- flash attention
// One block = 4 waves; each wave owns a 16-row Q tile of one head. K / V^T
// chunks are staged into LDS once per block (TDM or async b128).
__global__ void __launch_bounds__(128)
attn_kernel(const _Float16* __restrict__ qb, const _Float16* __restrict__ kb,
            const _Float16* __restrict__ vtb, const int* __restrict__ indices,
            _Float16* __restrict__ out) {
  __shared__ _Float16 kts[32][64];     // 32 keys x 64 d     (4 KB)
  __shared__ _Float16 vts[64][32];     // 64 d   x 32 keys   (4 KB)
  __shared__ _Float16 pbuf[4][16][32]; // per-wave P tiles   (4 KB)

  const int head = blockIdx.x >> 6;    // 16 heads x 64 q-blocks
  const int qblk = blockIdx.x & 63;
  const int wave = threadIdx.x >> 5;
  const int lane = threadIdx.x & 31;
  const int g = (lane >> 4) & 1;
  const int q0 = qblk * 64 + wave * 16;

  const _Float16* Qh  = qb  + (size_t)head * NTOK * HDIM;
  const _Float16* Kh  = kb  + (size_t)head * NTOK * HDIM;
  const _Float16* Vth = vtb + (size_t)head * HDIM * NTOK;

  const v16h aq0 = load_a_frag(Qh, HDIM, q0, 0);   // Q pre-scaled
  const v16h aq1 = load_a_frag(Qh, HDIM, q0, 32);

  v8f acc0 = {}, acc1 = {}, acc2 = {}, acc3 = {};
  float m[8], l[8];
#pragma unroll
  for (int r = 0; r < 8; ++r) { m[r] = -INFINITY; l[r] = 0.0f; }

  _Float16* pw = &pbuf[wave][0][0];

  for (int key0 = 0; key0 < NTOK; key0 += 32) {
    __syncthreads();                   // previous tile fully consumed
#if USE_TDM
    if (wave == 0) {
      // K chunk: 32 rows x 64 elems, rows HDIM apart (contiguous here)
      tdm_load_2d((unsigned)(uintptr_t)&kts[0][0],
                  Kh + (size_t)key0 * HDIM,
                  HDIM, 32, HDIM, 32, (unsigned long long)HDIM);
      // V^T chunk: 64 rows x 32 elems, rows NTOK apart
      tdm_load_2d((unsigned)(uintptr_t)&vts[0][0],
                  Vth + key0,
                  32, HDIM, 32, HDIM, (unsigned long long)NTOK);
      __builtin_amdgcn_s_wait_tensorcnt(0);
    }
#else
    {
      const _Float16* kg = Kh + (size_t)key0 * HDIM;
#pragma unroll
      for (int i = 0; i < 2; ++i) {
        const int c = threadIdx.x + 128 * i;        // 0..255 b128 chunks
        stage_b128(&kts[0][0] + c * 8, kg + c * 8);
        const int row = c >> 2;                     // V^T row (d)
        const int seg = c & 3;                      // 8-half segment
        stage_b128(&vts[row][seg * 8],
                   Vth + (size_t)row * NTOK + key0 + seg * 8);
      }
    }
    wait_async0();
#endif
    if (key0 + 32 < NTOK) {            // hint next chunk toward the caches
      __builtin_prefetch(Kh + (size_t)(key0 + 32) * HDIM + threadIdx.x * 16,
                         0, 1);
      __builtin_prefetch(Vth + (size_t)(threadIdx.x & 63) * NTOK + key0 + 32,
                         0, 1);
    }
    __syncthreads();                   // tile visible to all waves

    // ---- scores for 32 keys: two 16-key N tiles, d split 0..31 / 32..63
    v8f s0 = {}, s1 = {};
    s0 = wmma32(aq0, load_bT_frag(&kts[0][0], HDIM, 0, 0), s0);
    s0 = wmma32(aq1, load_bT_frag(&kts[0][0], HDIM, 0, 32), s0);
    s1 = wmma32(aq0, load_bT_frag(&kts[0][0], HDIM, 16, 0), s1);
    s1 = wmma32(aq1, load_bT_frag(&kts[0][0], HDIM, 16, 32), s1);

    // ---- online softmax (row = 16-lane half of one C VGPR)
    float alpha[8];
#pragma unroll
    for (int r = 0; r < 8; ++r) {
      const float x0 = s0[r];
      const float x1 = s1[r];
      float mx = fmaxf(x0, x1);
#pragma unroll
      for (int off = 8; off >= 1; off >>= 1)
        mx = fmaxf(mx, __shfl_xor(mx, off, 32));
      const float mn = fmaxf(m[r], mx);
      const float al = __expf(m[r] - mn);
      const float p0 = __expf(x0 - mn);
      const float p1 = __expf(x1 - mn);
      float ps = p0 + p1;
#pragma unroll
      for (int off = 8; off >= 1; off >>= 1)
        ps += __shfl_xor(ps, off, 32);
      l[r] = l[r] * al + ps;
      m[r] = mn;
      alpha[r] = al;
      // stage P (C layout -> row-major 16x32 tile) through per-wave LDS
      pbuf[wave][r + 8 * g][lane & 15]        = (_Float16)p0;
      pbuf[wave][r + 8 * g][16 + (lane & 15)] = (_Float16)p1;
    }
#pragma unroll
    for (int r = 0; r < 8; ++r) {
      acc0[r] *= alpha[r];
      acc1[r] *= alpha[r];
      acc2[r] *= alpha[r];
      acc3[r] *= alpha[r];
    }

    // ---- O += P x V : B-frag element(k,n) = V[key0+k][d0+n] = vts[d0+n][k]
    const v16h pa = load_a_frag(pw, 32, 0, 0);  // DS in-order within wave
    acc0 = wmma32(pa, load_bT_frag(&vts[0][0], 32, 0, 0), acc0);
    acc1 = wmma32(pa, load_bT_frag(&vts[0][0], 32, 16, 0), acc1);
    acc2 = wmma32(pa, load_bT_frag(&vts[0][0], 32, 32, 0), acc2);
    acc3 = wmma32(pa, load_bT_frag(&vts[0][0], 32, 48, 0), acc3);
  }

  // ---- epilogue: normalize and scatter back to original token order
#pragma unroll
  for (int r = 0; r < 8; ++r) {
    const int j = q0 + r + 8 * g;        // sorted position
    const int n = indices[j];            // original token
    const float li = 1.0f / l[r];
    const size_t base = (size_t)n * CDIM + head * HDIM + (lane & 15);
    out[base + 0]  = (_Float16)(acc0[r] * li);
    out[base + 16] = (_Float16)(acc1[r] * li);
    out[base + 32] = (_Float16)(acc2[r] * li);
    out[base + 48] = (_Float16)(acc3[r] * li);
  }
}

// ------------------------------------------------------- output projection
// y[n,c] = sum_d attn[n,d] * w_out[c,d]  (f32 output)
__global__ void __launch_bounds__(32)
out_gemm(const _Float16* __restrict__ ah, const _Float16* __restrict__ wh,
         float* __restrict__ y) {
  const int c0 = blockIdx.x * 16;
  const int t0 = blockIdx.y * 16;
  v8f acc = {};
#pragma unroll 2
  for (int kk = 0; kk < CDIM; kk += 32) {
    v16h a = load_a_frag(ah, CDIM, t0, kk);
    v16h b = load_bT_frag(wh, CDIM, c0, kk);
    acc = wmma32(a, b, acc);
  }
  const int lane = threadIdx.x & 31;
  const int g = (lane >> 4) & 1;
  const int c = c0 + (lane & 15);
#pragma unroll
  for (int r = 0; r < 8; ++r) {
    const int n = t0 + r + 8 * g;
    y[(size_t)n * CDIM + c] = acc[r];
  }
}

// -------------------------------------------------------------------- host
extern "C" void kernel_launch(void* const* d_in, const int* in_sizes, int n_in,
                              void* d_out, int out_size, void* d_ws,
                              size_t ws_size, hipStream_t stream) {
  (void)in_sizes; (void)n_in; (void)out_size; (void)ws_size;
  const float* x     = (const float*)d_in[0];
  const float* w_qkv = (const float*)d_in[1];
  const float* w_out = (const float*)d_in[2];
  const int*   idx   = (const int*)d_in[3];
  float* y = (float*)d_out;

  char* ws = (char*)d_ws;
  size_t off = 0;
  auto carve = [&](size_t bytes) {
    void* p = ws + off;
    off = (off + bytes + 255) & ~(size_t)255;
    return p;
  };
  _Float16* xh    = (_Float16*)carve((size_t)NTOK * CDIM * 2);           // 8 MB
  _Float16* wqkvh = (_Float16*)carve((size_t)QKV_DIM * CDIM * 2);        // 6 MB
  _Float16* wouth = (_Float16*)carve((size_t)CDIM * CDIM * 2);           // 2 MB
  _Float16* qh    = (_Float16*)carve((size_t)NHEADS * NTOK * HDIM * 2);  // 8 MB
  _Float16* kh    = (_Float16*)carve((size_t)NHEADS * NTOK * HDIM * 2);  // 8 MB
  _Float16* vth   = (_Float16*)carve((size_t)NHEADS * NTOK * HDIM * 2);  // 8 MB
  _Float16* atth  = (_Float16*)carve((size_t)NTOK * CDIM * 2);           // 8 MB
  int*      inv   = (int*)carve((size_t)NTOK * 4);

  { int n = NTOK * CDIM;
    cvt_f32_to_f16<<<(n + 255) / 256, 256, 0, stream>>>(x, xh, n); }
  { int n = QKV_DIM * CDIM;
    cvt_f32_to_f16<<<(n + 255) / 256, 256, 0, stream>>>(w_qkv, wqkvh, n); }
  { int n = CDIM * CDIM;
    cvt_f32_to_f16<<<(n + 255) / 256, 256, 0, stream>>>(w_out, wouth, n); }
  build_inv<<<NTOK / 256, 256, 0, stream>>>(idx, inv, NTOK);

  qkv_gemm<<<dim3(QKV_DIM / 16, NTOK / 16), 32, 0, stream>>>(xh, wqkvh, inv,
                                                             qh, kh, vth);
  attn_kernel<<<NHEADS * (NTOK / 64), 128, 0, stream>>>(qh, kh, vth, idx,
                                                        atth);
  out_gemm<<<dim3(CDIM / 16, NTOK / 16), 32, 0, stream>>>(atth, wouth, y);
}
